// NeuralAstar_31722628448272
// MI455X (gfx1250) — compile-verified, matrix operand order checked by reading, not run on hardware
//
#include <hip/hip_runtime.h>
#include <hip/hip_bf16.h>
#include <stdint.h>

typedef __attribute__((ext_vector_type(16))) _Float16 v16h;
typedef __attribute__((ext_vector_type(8)))  _Float16 v8h;
typedef __attribute__((ext_vector_type(8)))  float    v8f;

#define BB   32
#define HH   32
#define WW   32
#define HW   1024            // H*W
#define PP   32768           // B*H*W
#define TSTEPS 256           // int(0.25 * H * W)

// ---------------------------------------------------------------------------
// Prep: 2-channel NHWC encoder input (f16); weights -> [rs][Cout][Cin] f16
// ---------------------------------------------------------------------------
__global__ void build_x0_kernel(const float* __restrict__ mapd,
                                const float* __restrict__ startm,
                                const float* __restrict__ goalm,
                                _Float16* __restrict__ act) {  // [P][2]
    int p = blockIdx.x * blockDim.x + threadIdx.x;
    if (p >= PP) return;
    act[p * 2 + 0] = (_Float16)mapd[p];
    act[p * 2 + 1] = (_Float16)(startm[p] + goalm[p]);
}

__global__ void prepack_w_kernel(const float* __restrict__ w,   // [Cout][Cin][3][3]
                                 _Float16* __restrict__ wf,     // [9][Cout][Cin]
                                 int Cin, int Cout) {
    int i = blockIdx.x * blockDim.x + threadIdx.x;
    if (i >= Cout * Cin * 9) return;
    int rs = i % 9;
    int t  = i / 9;
    int ci = t % Cin;
    int co = t / Cin;
    wf[((size_t)rs * Cout + co) * Cin + ci] = (_Float16)w[i];
}

// ---------------------------------------------------------------------------
// Layer 0 (CIN=2): direct implicit-GEMM, masked scalar fragments (K=18 < 32).
// ---------------------------------------------------------------------------
__global__ __launch_bounds__(256)
void conv_wmma_l0_kernel(const _Float16* __restrict__ act,   // [P][2]
                         const _Float16* __restrict__ wgt,   // [9][32][2]
                         const float*    __restrict__ bias,  // [32]
                         float*          __restrict__ out) { // [32][P]
    constexpr int CIN = 2, COUT = 32;
    const int lane   = threadIdx.x & 31;
    const int waveId = threadIdx.x >> 5;
    constexpr int pixTiles = PP / 16;
    const int tile = blockIdx.x * (blockDim.x >> 5) + waveId;
    if (tile >= pixTiles * (COUT / 16)) return;
    const int pixTile  = tile % pixTiles;
    const int coutTile = tile / pixTiles;
    const int half = lane >> 4;
    const int l16  = lane & 15;
    const int p  = pixTile * 16 + l16;
    const int py = (p >> 5) & 31;
    const int px = p & 31;
    const int m  = coutTile * 16 + l16;

    v8f c = {0.f, 0.f, 0.f, 0.f, 0.f, 0.f, 0.f, 0.f};
    #pragma unroll
    for (int r = -1; r <= 1; ++r) {
        #pragma unroll
        for (int s = -1; s <= 1; ++s) {
            const int rs = (r + 1) * 3 + (s + 1);
            const int yy = py + r, xx = px + s;
            const bool inb = ((unsigned)yy < 32u) && ((unsigned)xx < 32u);
            const _Float16* arow  = act + (size_t)(p + r * 32 + s) * CIN;
            const _Float16* wbase = wgt + ((size_t)rs * COUT + m) * CIN;
            v16h a, b;
            #pragma unroll
            for (int j = 0; j < 16; ++j) {
                int ko = (j < 8 ? j : j + 8) + half * 8;
                _Float16 av = (_Float16)0.f, bv = (_Float16)0.f;
                if (ko < CIN) {
                    av = wbase[ko];
                    if (inb) bv = arow[ko];
                }
                a[j] = av; b[j] = bv;
            }
            c = __builtin_amdgcn_wmma_f32_16x16x32_f16(
                    false, a, false, b, (short)0, c, false, false);
        }
    }
    #pragma unroll
    for (int rr = 0; rr < 8; ++rr) {
        int mm = coutTile * 16 + rr + half * 8;
        out[(size_t)mm * PP + pixTile * 16 + l16] = c[rr] + bias[mm];
    }
}

// ---------------------------------------------------------------------------
// LDS-staged implicit-GEMM conv (CIN >= 32).  One block = 4-row stripe of one
// image (128 pixels, 8 waves = 8 pixel tiles).  Halo rows y0-1..y0+4 staged
// once via GLOBAL_LOAD_ASYNC_TO_LDS_B128 (ASYNCcnt, no VGPR round-trip) and
// reused by all COUT/16 cout tiles; B fragments come from ds_load_b128,
// A fragments stream from L2-resident prepacked weights (prefetched ahead).
// ---------------------------------------------------------------------------
template<int CIN, int COUT>
__global__ __launch_bounds__(256)
void conv_wmma_lds_kernel(const _Float16* __restrict__ act,   // [P][CIN]
                          const _Float16* __restrict__ wgt,   // [9][COUT][CIN]
                          const float*    __restrict__ bias,  // [COUT]
                          float*          __restrict__ out) { // [COUT][P]
    __shared__ __align__(16) _Float16 sact[6 * 32 * CIN];

    const int tid    = threadIdx.x;
    const int lane   = tid & 31;
    const int waveId = tid >> 5;
    const int blk = blockIdx.x;          // 0..255
    const int b   = blk >> 3;            // batch image
    const int y0  = (blk & 7) * 4;       // stripe top row

    // ---- async halo fill: rows y0-1 .. y0+4, 16B chunks ----
    constexpr int C8     = CIN / 8;
    constexpr int CHUNKS = 6 * 32 * C8;  // multiple of 256
    for (int i = tid; i < CHUNKS; i += 256) {
        int c8  = i % C8;
        int pxr = i / C8;                // row*32 + px, row in 0..5
        int y   = y0 - 1 + (pxr >> 5);
        _Float16* dst = &sact[(size_t)pxr * CIN + c8 * 8];
        if ((unsigned)y < 32u) {
            const _Float16* src = act
                + (size_t)((b * 32 + y) * 32 + (pxr & 31)) * CIN + c8 * 8;
            // LDS aperture: low 32 bits of a generic __shared__ address are
            // the LDS byte offset the async instruction's VDST expects.
            uint32_t ldsa = (uint32_t)(uintptr_t)dst;
            uint64_t ga   = (uint64_t)(uintptr_t)src;
            asm volatile("global_load_async_to_lds_b128 %0, %1, off"
                         :: "v"(ldsa), "v"(ga) : "memory");
        } else {
            v8h z = {};
            *(v8h*)dst = z;              // zero-fill out-of-image halo rows
        }
    }
    asm volatile("s_wait_asynccnt 0x0" ::: "memory");
    __syncthreads();

    const int half = lane >> 4;          // ISA f16 fragment K-group select
    const int l16  = lane & 15;
    const int lp   = waveId * 16 + l16;  // local pixel 0..127
    const int ly   = lp >> 5, lx = lp & 31;
    const int gp   = (b * 32 + y0 + ly) * 32 + lx;   // global pixel (store col)

    for (int ct = 0; ct < COUT / 16; ++ct) {
        if (ct + 1 < COUT / 16)          // prefetch next cout tile's weights
            __builtin_prefetch(wgt + (size_t)((ct + 1) * 16 + l16) * CIN, 0, 1);

        v8f c = {0.f, 0.f, 0.f, 0.f, 0.f, 0.f, 0.f, 0.f};
        const _Float16* wct = wgt + (size_t)(ct * 16 + l16) * CIN;

        #pragma unroll
        for (int r = -1; r <= 1; ++r) {
            #pragma unroll
            for (int s = -1; s <= 1; ++s) {
                const int rs   = (r + 1) * 3 + (s + 1);
                const int scol = lx + s;
                const bool inb = (unsigned)scol < 32u;   // x border only
                const _Float16* wbase = wct + (size_t)rs * COUT * CIN;
                #pragma unroll
                for (int k0 = 0; k0 < CIN; k0 += 32) {
                    const int ca = k0 + half * 8;        // K {0..7}/{8..15}
                    const int cb = k0 + 16 + half * 8;   // K {16..23}/{24..31}
                    v8h alo = *(const v8h*)(wbase + ca);
                    v8h ahi = *(const v8h*)(wbase + cb);
                    v8h blo = {}, bhi = {};
                    if (inb) {
                        const _Float16* arow =
                            &sact[(size_t)(((ly + 1 + r) * 32 + scol)) * CIN];
                        blo = *(const v8h*)(arow + ca);   // ds_load_b128
                        bhi = *(const v8h*)(arow + cb);
                    }
                    v16h a = __builtin_shufflevector(alo, ahi,
                              0,1,2,3,4,5,6,7,8,9,10,11,12,13,14,15);
                    v16h bfrag = __builtin_shufflevector(blo, bhi,
                              0,1,2,3,4,5,6,7,8,9,10,11,12,13,14,15);
                    c = __builtin_amdgcn_wmma_f32_16x16x32_f16(
                            false, a, false, bfrag, (short)0, c, false, false);
                }
            }
        }

        #pragma unroll
        for (int rr = 0; rr < 8; ++rr) {
            int mm = ct * 16 + rr + half * 8;
            out[(size_t)mm * PP + gp] = c[rr] + bias[mm];
        }
    }
}

// Last layer: Cout = 1 -> plain f32 dot product (WMMA would waste 15/16 of M).
__global__ void conv_last_kernel(const _Float16* __restrict__ act, // [P][256]
                                 const float* __restrict__ wgt,    // [1][256][3][3]
                                 const float* __restrict__ bias,
                                 float* __restrict__ out) {
    int p = blockIdx.x * blockDim.x + threadIdx.x;
    if (p >= PP) return;
    int py = (p >> 5) & 31, px = p & 31;
    float acc = 0.f;
    #pragma unroll
    for (int r = -1; r <= 1; ++r)
        #pragma unroll
        for (int s = -1; s <= 1; ++s) {
            int yy = py + r, xx = px + s;
            if ((unsigned)yy >= 32u || (unsigned)xx >= 32u) continue;
            const _Float16* arow = act + (size_t)(p + r * 32 + s) * 256;
            int rs = (r + 1) * 3 + (s + 1);
            for (int ci = 0; ci < 256; ++ci)
                acc += wgt[ci * 9 + rs] * (float)arow[ci];
        }
    out[p] = acc + bias[0];
}

// ---------------------------------------------------------------------------
// BatchNorm (training-mode batch stats) helpers
// ---------------------------------------------------------------------------
__global__ void bn_stats_kernel(const float* __restrict__ y,   // [C][P]
                                float* __restrict__ mean,
                                float* __restrict__ istd) {
    int c = blockIdx.x;
    __shared__ float ssum[256], ssq[256];
    float s = 0.f, q = 0.f;
    for (int i = threadIdx.x; i < PP; i += blockDim.x) {
        float v = y[(size_t)c * PP + i];
        s += v; q += v * v;
    }
    ssum[threadIdx.x] = s; ssq[threadIdx.x] = q;
    __syncthreads();
    for (int off = 128; off; off >>= 1) {
        if (threadIdx.x < off) {
            ssum[threadIdx.x] += ssum[threadIdx.x + off];
            ssq[threadIdx.x]  += ssq[threadIdx.x + off];
        }
        __syncthreads();
    }
    if (threadIdx.x == 0) {
        float mu  = ssum[0] * (1.f / PP);
        float var = ssq[0] * (1.f / PP) - mu * mu;
        mean[c] = mu;
        istd[c] = rsqrtf(var + 1e-5f);
    }
}

__global__ void bn_relu_f16_kernel(const float* __restrict__ y,   // [C][P]
                                   const float* __restrict__ mean,
                                   const float* __restrict__ istd,
                                   const float* __restrict__ gamma,
                                   const float* __restrict__ beta,
                                   _Float16* __restrict__ act,    // [P][C] NHWC
                                   int C) {
    size_t i = (size_t)blockIdx.x * blockDim.x + threadIdx.x;
    if (i >= (size_t)C * PP) return;
    int c = (int)(i / PP);
    int p = (int)(i - (size_t)c * PP);
    float v = (y[i] - mean[c]) * istd[c] * gamma[c] + beta[c];
    act[(size_t)p * C + c] = (_Float16)(v > 0.f ? v : 0.f);
}

__global__ void bn_sigmoid_kernel(const float* __restrict__ y,
                                  const float* __restrict__ mean,
                                  const float* __restrict__ istd,
                                  const float* __restrict__ gamma,
                                  const float* __restrict__ beta,
                                  float* __restrict__ cost) {
    int i = blockIdx.x * blockDim.x + threadIdx.x;
    if (i >= PP) return;
    float v = (y[i] - mean[0]) * istd[0] * gamma[0] + beta[0];
    cost[i] = 1.f / (1.f + __expf(-v));
}

// ---------------------------------------------------------------------------
// Differentiable A* forward: one workgroup, 32 wave32 waves = one wave/batch.
// ---------------------------------------------------------------------------
__global__ __launch_bounds__(1024)
void astar_kernel(const float* __restrict__ cost,
                  const float* __restrict__ startm,
                  const float* __restrict__ goalm,
                  const float* __restrict__ mapd,
                  float* __restrict__ g,
                  float* __restrict__ openm,
                  float* __restrict__ hist,
                  int*   __restrict__ parents,
                  float* __restrict__ htot,
                  float* __restrict__ out_hist,
                  float* __restrict__ out_path) {
    const int lane = threadIdx.x & 31;
    const int b    = threadIdx.x >> 5;
    const float inv_scale = 1.0f / 5.656854249492380f;  // 1/sqrt(W)

    __shared__ int s_goal[32];
    __shared__ int s_any;
    __shared__ int s_active;
    __shared__ int s_tcnt;

    const float* cst = cost  + b * HW;
    const float* gl  = goalm + b * HW;
    float* gb  = g       + b * HW;
    float* ob  = openm   + b * HW;
    float* hb  = hist    + b * HW;
    float* ht  = htot    + b * HW;
    int*   par = parents + b * HW;

    int gidx = -1;
    for (int j = 0; j < 32; ++j) {
        int c = lane + 32 * j;
        if (gl[c] > 0.5f) gidx = c;
    }
    for (int off = 16; off; off >>= 1) {
        int o = __shfl_xor(gidx, off, 32);
        if (o > gidx) gidx = o;
    }
    if (lane == 0) s_goal[b] = gidx;
    const int gy = gidx >> 5, gx = gidx & 31;

    for (int j = 0; j < 32; ++j) {
        int c = lane + 32 * j;
        int y = c >> 5, x = c & 31;
        float dy = fabsf((float)(y - gy));
        float dx = fabsf((float)(x - gx));
        float cheb = dy + dx - fminf(dy, dx);
        float euc  = sqrtf(dy * dy + dx * dx);
        ht[c]  = cheb + 0.001f * euc + cst[c];
        gb[c]  = 0.f;
        ob[c]  = startm[b * HW + c];
        hb[c]  = 0.f;
        par[c] = gidx;
    }
    if (threadIdx.x == 0) { s_active = 1; s_tcnt = 0; }
    __syncthreads();

    for (int t = 0; t < TSTEPS; ++t) {
        if (threadIdx.x == 0) s_any = 0;
        __syncthreads();
        if (s_active) {
            float best = -1.f; int bidx = 0;
            for (int j = 0; j < 32; ++j) {
                int c = lane + 32 * j;
                float f  = 0.5f * gb[c] + 0.5f * ht[c];
                float fe = __expf(-f * inv_scale) * ob[c];
                if (fe > best) { best = fe; bidx = c; }
            }
            for (int off = 16; off; off >>= 1) {
                float of = __shfl_xor(best, off, 32);
                int   oi = __shfl_xor(bidx, off, 32);
                if (of > best || (of == best && oi < bidx)) { best = of; bidx = oi; }
            }
            if (lane == 0) {
                int sel = bidx;
                int unsolved = (sel != s_goal[b]) ? 1 : 0;
                float g2 = gb[sel] + cst[sel];
                hb[sel] = 1.f;
                if (unsolved) { ob[sel] = 0.f; s_any = 1; }
                int sy = sel >> 5, sx = sel & 31;
                #pragma unroll
                for (int r = -1; r <= 1; ++r)
                    #pragma unroll
                    for (int s = -1; s <= 1; ++s) {
                        if (r == 0 && s == 0) continue;
                        int yy = sy + r, xx = sx + s;
                        if ((unsigned)yy >= 32u || (unsigned)xx >= 32u) continue;
                        int n = (yy << 5) | xx;
                        if (mapd[b * HW + n] <= 0.5f) continue;
                        float op = ob[n], hi = hb[n], gn = gb[n];
                        float idxv = (1.f - op) * (1.f - hi) +
                                     op * ((gn > g2) ? 1.f : 0.f);
                        if (idxv > 0.5f) {
                            gb[n]  = g2;
                            ob[n]  = 1.f;
                            par[n] = sel;
                        }
                    }
            }
        }
        __syncthreads();
        if (threadIdx.x == 0) {
            if (s_active) { s_tcnt++; if (!s_any) s_active = 0; }
        }
        __syncthreads();
    }

    for (int j = 0; j < 32; ++j) {
        int c = lane + 32 * j;
        out_hist[b * HW + c] = hb[c];
        out_path[b * HW + c] = (c == s_goal[b]) ? 1.f : 0.f;
    }
    if (lane == 0) {
        int loc = par[s_goal[b]];
        int ct  = s_tcnt - 1;
        for (int i = 0; i < TSTEPS; ++i) {
            if (i < ct) {
                out_path[b * HW + loc] = 1.f;
                loc = par[loc];
            }
        }
    }
}

// ---------------------------------------------------------------------------
// Host-side orchestration
// ---------------------------------------------------------------------------
extern "C" void kernel_launch(void* const* d_in, const int* in_sizes, int n_in,
                              void* d_out, int out_size, void* d_ws, size_t ws_size,
                              hipStream_t stream) {
    (void)in_sizes; (void)n_in; (void)out_size; (void)ws_size;

    const float* mapd   = (const float*)d_in[0];
    const float* startm = (const float*)d_in[1];
    const float* goalm  = (const float*)d_in[2];
    const float *Wt[5], *Bs[5], *Gm[5], *Bt[5];
    for (int i = 0; i < 5; ++i) {
        Wt[i] = (const float*)d_in[3 + 4 * i];
        Bs[i] = (const float*)d_in[4 + 4 * i];
        Gm[i] = (const float*)d_in[5 + 4 * i];
        Bt[i] = (const float*)d_in[6 + 4 * i];
    }

    char* ws = (char*)d_ws;
    size_t off = 0;
    auto take = [&](size_t bytes) -> char* {
        char* p = ws + off;
        off += (bytes + 255) & ~(size_t)255;
        return p;
    };
    const int Cin[4]  = { 2, 32, 64, 128 };
    const int Cout[4] = { 32, 64, 128, 256 };
    const int wsz[4]  = { 32 * 18, 64 * 288, 128 * 576, 256 * 1152 };
    _Float16* wf[4];
    for (int i = 0; i < 4; ++i) wf[i] = (_Float16*)take((size_t)wsz[i] * 2);
    _Float16* actA = (_Float16*)take((size_t)256 * PP * 2);
    _Float16* actB = (_Float16*)take((size_t)256 * PP * 2);
    float* raw   = (float*)take((size_t)256 * PP * 4);
    float* mean  = (float*)take(256 * 4);
    float* istd  = (float*)take(256 * 4);
    float* cost  = (float*)take((size_t)PP * 4);
    float* gbuf  = (float*)take((size_t)PP * 4);
    float* obuf  = (float*)take((size_t)PP * 4);
    float* hbuf  = (float*)take((size_t)PP * 4);
    float* htot  = (float*)take((size_t)PP * 4);
    int*   parb  = (int*)  take((size_t)PP * 4);

    // 1) inputs + weight prepack
    build_x0_kernel<<<PP / 256, 256, 0, stream>>>(mapd, startm, goalm, actA);
    for (int i = 0; i < 4; ++i)
        prepack_w_kernel<<<(wsz[i] + 255) / 256, 256, 0, stream>>>(
            Wt[i], wf[i], Cin[i], Cout[i]);

    // 2) WMMA conv layers + training-mode BN + ReLU (NHWC ping-pong)
    _Float16* cur = actA;
    _Float16* nxt = actB;
    for (int i = 0; i < 4; ++i) {
        if (i == 0) {
            int tiles = (PP / 16) * (Cout[0] / 16);
            conv_wmma_l0_kernel<<<tiles / 8, 256, 0, stream>>>(cur, wf[0], Bs[0], raw);
        } else if (i == 1) {
            conv_wmma_lds_kernel<32,  64><<<256, 256, 0, stream>>>(cur, wf[1], Bs[1], raw);
        } else if (i == 2) {
            conv_wmma_lds_kernel<64, 128><<<256, 256, 0, stream>>>(cur, wf[2], Bs[2], raw);
        } else {
            conv_wmma_lds_kernel<128,256><<<256, 256, 0, stream>>>(cur, wf[3], Bs[3], raw);
        }
        bn_stats_kernel<<<Cout[i], 256, 0, stream>>>(raw, mean, istd);
        int n = Cout[i] * PP;
        bn_relu_f16_kernel<<<n / 256, 256, 0, stream>>>(raw, mean, istd,
                                                        Gm[i], Bt[i], nxt, Cout[i]);
        _Float16* tmp = cur; cur = nxt; nxt = tmp;
    }

    // 3) last conv (Cout=1) + BN + sigmoid -> cost maps
    conv_last_kernel<<<PP / 256, 256, 0, stream>>>(cur, Wt[4], Bs[4], raw);
    bn_stats_kernel<<<1, 256, 0, stream>>>(raw, mean, istd);
    bn_sigmoid_kernel<<<PP / 256, 256, 0, stream>>>(raw, mean, istd, Gm[4], Bt[4], cost);

    // 4) differentiable A* scan + backtrack -> (hist, path)
    astar_kernel<<<1, 1024, 0, stream>>>(cost, startm, goalm, mapd,
                                         gbuf, obuf, hbuf, parb, htot,
                                         (float*)d_out, (float*)d_out + PP);
}
